// event_network_730144440408
// MI455X (gfx1250) — compile-verified, matrix operand order checked by reading, not run on hardware
//
#include <hip/hip_runtime.h>
#include <hip/hip_bf16.h>

typedef unsigned short u16;
typedef __attribute__((ext_vector_type(16))) __bf16 v16bf;
typedef __attribute__((ext_vector_type(8)))  float  v8f;

union Frag { v16bf v; uint4 q[2]; };

#define BROWS 8192
#define KD 512
#define HD 512
#define MT 64
#define LDA 520   // bf16 elems; 1040 B = 65*16 (aligned), 260 dwords -> 4-bank row advance
#define LDB 40    // bf16 elems; 80 B = 5*16 (aligned), 20 dwords -> conflict-free 16-row tiling
#define BTSZ (128 * LDB)   // one panel buffer, in u16 elems (10240 B)

#define SM_A     0            // 64*520*2  = 66560
#define SM_H     66560        // 64*520*2  = 66560
#define SM_BT0   133120       // 3 panel buffers, contiguous, 10240 B each
#define SM_WP    163840       // 16*520*2  = 16640
#define SM_LOG   180480       // 64*9*4    = 2304
#define SM_TOTAL 182784

static __device__ __forceinline__ u16 f32_to_bf16_bits(float f) {
  union { float f; unsigned u; } cv; cv.f = f;
  unsigned u = cv.u;
  u += 0x7fffu + ((u >> 16) & 1u);   // round-to-nearest-even
  return (u16)(u >> 16);
}

// gfx1250 async global->LDS DMA (ASYNCcnt); lds_off = wave-relative LDS byte addr
static __device__ __forceinline__ void async_ld128(unsigned lds_off, const void* g) {
  asm volatile("global_load_async_to_lds_b128 %0, %1, off"
               :: "v"(lds_off), "v"(g) : "memory");
}
static __device__ __forceinline__ void wait_asynccnt0() {
  asm volatile("s_wait_asynccnt 0" ::: "memory");
}
// allow 2 async ops (one staged panel) to remain in flight
static __device__ __forceinline__ void wait_asynccnt2() {
  asm volatile("s_wait_asynccnt 2" ::: "memory");
}

// ---------------- conversion kernels ----------------

__global__ __launch_bounds__(256) void cvt_inp_kernel(const float* __restrict__ in,
                                                      u16* __restrict__ out) {
  size_t i = ((size_t)blockIdx.x * 256 + threadIdx.x) * 4;
  float4 v = *(const float4*)(in + i);
  alignas(8) u16 o[4] = {f32_to_bf16_bits(v.x), f32_to_bf16_bits(v.y),
                         f32_to_bf16_bits(v.z), f32_to_bf16_bits(v.w)};
  *(uint2*)(out + i) = *(const uint2*)o;
}

// Wx [net][k=512][j=512] f32  ->  Wt [net][j=512][k=512] bf16 (LDS-tiled transpose)
__global__ __launch_bounds__(256) void cvt_wx_kernel(const float* __restrict__ W,
                                                     u16* __restrict__ Wt) {
  __shared__ float tile[32][33];
  const float* Wn = W  + (size_t)blockIdx.z * KD * HD;
  u16*        Wtn = Wt + (size_t)blockIdx.z * KD * HD;
  int k0 = blockIdx.y * 32;
  int j0 = blockIdx.x * 32;
#pragma unroll
  for (int i = 0; i < 4; ++i)
    tile[threadIdx.y + i * 8][threadIdx.x] =
        Wn[(size_t)(k0 + threadIdx.y + i * 8) * HD + j0 + threadIdx.x];
  __syncthreads();
#pragma unroll
  for (int i = 0; i < 4; ++i)
    Wtn[(size_t)(j0 + threadIdx.y + i * 8) * KD + k0 + threadIdx.x] =
        f32_to_bf16_bits(tile[threadIdx.x][threadIdx.y + i * 8]);
}

// Wp [net][k=512][j=8] f32 -> Wpt [net][j=8][k=512] bf16  (tiny)
__global__ __launch_bounds__(256) void cvt_wp_kernel(const float* __restrict__ Wp,
                                                     u16* __restrict__ Wpt) {
  const float* s = Wp  + (size_t)blockIdx.x * KD * 8;
  u16*         d = Wpt + (size_t)blockIdx.x * 8 * KD;
#pragma unroll
  for (int i = 0; i < 16; ++i) {
    int o = threadIdx.x + i * 256;     // 0..4095
    int j = o >> 9, k = o & 511;
    d[o] = f32_to_bf16_bits(s[(size_t)k * 8 + j]);
  }
}

// got_event = sigmoid(inp @ We + be); one wave32 per batch row
__global__ __launch_bounds__(256) void ge_kernel(const float* __restrict__ inp,
                                                 const float* __restrict__ We,
                                                 const float* __restrict__ be,
                                                 float* __restrict__ ge) {
  int lane = threadIdx.x & 31;
  int wave = threadIdx.x >> 5;
  int row = blockIdx.x * 8 + wave;
  const float* ip = inp + (size_t)row * KD;
  float s = 0.f;
#pragma unroll
  for (int k = 0; k < KD; k += 32) s = fmaf(ip[k + lane], We[k + lane], s);
#pragma unroll
  for (int off = 16; off > 0; off >>= 1) s += __shfl_xor(s, off, 32);
  if (lane == 0) ge[row] = 1.0f / (1.0f + __expf(-(s + be[0])));
}

// ---------------- fused level kernel ----------------
// grid = (BROWS/MT, nnets), block = 256 (8 wave32s)

// stage weight panel g (g = nb*16 + kp): Wt[nb*128 .. +127][kp*32 .. +31]
static __device__ __forceinline__ void stage_bt_async(const u16* __restrict__ WtN,
                                                      u16* buf, int g, int tid) {
#pragma unroll
  for (int i = 0; i < 2; ++i) {
    int idx = tid + i * 256;
    int j = idx >> 2, c = idx & 3;
    async_ld128((unsigned)(size_t)(buf + j * LDB + c * 8),
                WtN + (size_t)((g >> 4) * 128 + j) * KD + (g & 15) * 32 + c * 8);
  }
}

__global__ __launch_bounds__(256)
void tree_level_kernel(const u16* __restrict__ inp16,
                       const u16* __restrict__ Wt,
                       const float* __restrict__ bx,
                       const u16* __restrict__ Wpt,
                       const float* __restrict__ bp,
                       const float* __restrict__ parent_in,   // null -> use ge
                       const float* __restrict__ ge,
                       float* __restrict__ out,               // row stride 8*nnets+1
                       float* __restrict__ parent_out,        // null on last level
                       int nnets) {
  extern __shared__ char smem[];
  u16*   sA     = (u16*)(smem + SM_A);
  u16*   sH     = (u16*)(smem + SM_H);
  u16*   sWp    = (u16*)(smem + SM_WP);
  float* sLog   = (float*)(smem + SM_LOG);
  u16*   btbase = (u16*)(smem + SM_BT0);   // 3 contiguous panel buffers of BTSZ u16

  const int tid  = threadIdx.x;
  const int lane = tid & 31;
  const int wave = tid >> 5;
  const int half = lane >> 4;    // WMMA lane half (K-chunk select)
  const int l16  = lane & 15;    // WMMA M/N index within tile
  const int row0 = blockIdx.x * MT;
  const int net  = blockIdx.y;

  const u16* WtN  = Wt  + (size_t)net * HD * KD;
  const u16* WptN = Wpt + (size_t)net * 8 * KD;

  // Async-stage A tile (64 x 512 bf16) and WpT (rows 8..15 zero-padded via DS stores)
#pragma unroll
  for (int i = 0; i < 16; ++i) {
    int idx = tid + i * 256;
    int r = idx >> 6, c = idx & 63;
    async_ld128((unsigned)(size_t)(sA + r * LDA + c * 8),
                inp16 + (size_t)(row0 + r) * KD + c * 8);
  }
#pragma unroll
  for (int i = 0; i < 4; ++i) {
    int idx = tid + i * 256;
    int r = idx >> 6, c = idx & 63;
    if (r < 8)
      async_ld128((unsigned)(size_t)(sWp + r * LDA + c * 8),
                  WptN + (size_t)r * KD + c * 8);
    else
      *(uint4*)(sWp + r * LDA + c * 8) = make_uint4(0u, 0u, 0u, 0u);
  }

  const int mw = wave & 1;   // wave tile: 32x32 inside 64x128 block tile
  const int nw = wave >> 1;

  // Software pipeline over 64 global panels g = nb*16 + kp, 2 panels in flight.
  stage_bt_async(WtN, btbase + 0 * BTSZ, 0, tid);
  stage_bt_async(WtN, btbase + 1 * BTSZ, 1, tid);

  for (int nb = 0; nb < 4; ++nb) {      // 4 x 128-column blocks of the 512 hidden
    v8f acc[2][2] = {};
    for (int kp = 0; kp < 16; ++kp) {   // K panels of 32
      const int g = nb * 16 + kp;
      if (g == 63) wait_asynccnt0();    // tail: drain last panel
      else         wait_asynccnt2();    // drain panel g, keep panel g+1 in flight
      __syncthreads();                  // everyone's panel-g data visible
      if (g + 2 < 64)
        stage_bt_async(WtN, btbase + (unsigned)((g + 2) % 3) * BTSZ, g + 2, tid);

      const u16* btp = btbase + (unsigned)(g % 3) * BTSZ;
      Frag a[2], b[2];
#pragma unroll
      for (int i = 0; i < 2; ++i) {     // A frag: lane row = l16, K chunks half*8 / 16+half*8
        const u16* p = sA + (mw * 32 + i * 16 + l16) * LDA + kp * 32 + half * 8;
        a[i].q[0] = *(const uint4*)(p);
        a[i].q[1] = *(const uint4*)(p + 16);
      }
#pragma unroll
      for (int j = 0; j < 2; ++j) {     // B frag from K-major staged weights
        const u16* p = btp + (nw * 32 + j * 16 + l16) * LDB + half * 8;
        b[j].q[0] = *(const uint4*)(p);
        b[j].q[1] = *(const uint4*)(p + 16);
      }
#pragma unroll
      for (int i = 0; i < 2; ++i)
#pragma unroll
        for (int j = 0; j < 2; ++j)
          acc[i][j] = __builtin_amdgcn_wmma_f32_16x16x32_bf16(
              false, a[i].v, false, b[j].v, (short)0, acc[i][j], false, false);
    }
    // bias + ReLU, store H block to LDS as bf16 (C/D layout: m = r + 8*half, n = l16)
#pragma unroll
    for (int j = 0; j < 2; ++j) {
      int ng = nb * 128 + nw * 32 + j * 16 + l16;
      float bias = bx[(size_t)net * HD + ng];
#pragma unroll
      for (int i = 0; i < 2; ++i) {
        const float* ap = (const float*)&acc[i][j];
#pragma unroll
        for (int r = 0; r < 8; ++r) {
          int m = mw * 32 + i * 16 + 8 * half + r;
          float h = fmaxf(ap[r] + bias, 0.0f);
          sH[m * LDA + ng] = f32_to_bf16_bits(h);
        }
      }
    }
  }
  __syncthreads();

  // Phase 2: logits[64][8] = H @ WpT via WMMA (waves 0..3, one 16-row tile each)
  if (wave < 4) {
    v8f acc = {};
    for (int kp = 0; kp < 16; ++kp) {
      Frag a, b;
      const u16* pa = sH + (wave * 16 + l16) * LDA + kp * 32 + half * 8;
      a.q[0] = *(const uint4*)(pa);
      a.q[1] = *(const uint4*)(pa + 16);
      const u16* pb = sWp + l16 * LDA + kp * 32 + half * 8;
      b.q[0] = *(const uint4*)(pb);
      b.q[1] = *(const uint4*)(pb + 16);
      acc = __builtin_amdgcn_wmma_f32_16x16x32_bf16(
          false, a.v, false, b.v, (short)0, acc, false, false);
    }
    if (l16 < 8) {
      float bias = bp[(size_t)net * 8 + l16];
      const float* ap = (const float*)&acc;
#pragma unroll
      for (int r = 0; r < 8; ++r)
        sLog[(wave * 16 + 8 * half + r) * 9 + l16] = ap[r] + bias;
    }
  }
  __syncthreads();

  // Phase 3: softmax over 8, scale by parent prob, write outputs
  if (tid < 64) {
    int brow = row0 + tid;
    float v[8], mx = -3.4e38f;
#pragma unroll
    for (int j = 0; j < 8; ++j) { v[j] = sLog[tid * 9 + j]; mx = fmaxf(mx, v[j]); }
    float s = 0.f;
#pragma unroll
    for (int j = 0; j < 8; ++j) { v[j] = __expf(v[j] - mx); s += v[j]; }
    float gev = ge[brow];
    float pscale = parent_in ? parent_in[(size_t)brow * nnets + net] : gev;
    float scale = pscale / s;
    size_t ob = (size_t)brow * (nnets * 8 + 1) + net * 8;
#pragma unroll
    for (int j = 0; j < 8; ++j) {
      float p = v[j] * scale;
      out[ob + j] = p;
      if (parent_out) parent_out[(size_t)brow * nnets * 8 + net * 8 + j] = p;
    }
    if (net == 0) out[(size_t)brow * (nnets * 8 + 1) + nnets * 8] = 1.0f - gev;
  }
}

// ---------------- host launcher ----------------

extern "C" void kernel_launch(void* const* d_in, const int* in_sizes, int n_in,
                              void* d_out, int out_size, void* d_ws, size_t ws_size,
                              hipStream_t stream) {
  (void)in_sizes; (void)n_in; (void)out_size; (void)ws_size;
  const float* inp = (const float*)d_in[0];
  const float* Wx[3] = {(const float*)d_in[1], (const float*)d_in[5], (const float*)d_in[9]};
  const float* bx[3] = {(const float*)d_in[2], (const float*)d_in[6], (const float*)d_in[10]};
  const float* Wp[3] = {(const float*)d_in[3], (const float*)d_in[7], (const float*)d_in[11]};
  const float* bp[3] = {(const float*)d_in[4], (const float*)d_in[8], (const float*)d_in[12]};
  const float* We = (const float*)d_in[13];
  const float* be = (const float*)d_in[14];

  char* ws = (char*)d_ws;
  u16*   inp16  = (u16*)(ws + 0);                       //  8 MB
  u16*   wt[3]  = {(u16*)(ws + 8388608),                //  0.5 MB
                   (u16*)(ws + 8912896),                //  4 MB
                   (u16*)(ws + 13107200)};              // 32 MB
  u16*   wpt[3] = {(u16*)(ws + 46661632),
                   (u16*)(ws + 46669824),
                   (u16*)(ws + 46735360)};
  float* gebuf  = (float*)(ws + 47259648);
  float* par0   = (float*)(ws + 47292416);              // [8192][8]
  float* par1   = (float*)(ws + 47554560);              // [8192][64]

  float* out0 = (float*)d_out;                          // [8192][9]
  float* out1 = out0 + (size_t)8192 * 9;                // [8192][65]
  float* out2 = out1 + (size_t)8192 * 65;               // [8192][513]

  (void)hipFuncSetAttribute((const void*)tree_level_kernel,
                            hipFuncAttributeMaxDynamicSharedMemorySize, SM_TOTAL);

  cvt_inp_kernel<<<(BROWS * KD) / (256 * 4), 256, 0, stream>>>(inp, inp16);
  const int nets[3] = {1, 8, 64};
  for (int l = 0; l < 3; ++l) {
    cvt_wx_kernel<<<dim3(16, 16, nets[l]), dim3(32, 8), 0, stream>>>(Wx[l], wt[l]);
    cvt_wp_kernel<<<nets[l], 256, 0, stream>>>(Wp[l], wpt[l]);
  }
  ge_kernel<<<BROWS / 8, 256, 0, stream>>>(inp, We, be, gebuf);

  tree_level_kernel<<<dim3(BROWS / MT, 1), 256, SM_TOTAL, stream>>>(
      inp16, wt[0], bx[0], wpt[0], bp[0], nullptr, gebuf, out0, par0, 1);
  tree_level_kernel<<<dim3(BROWS / MT, 8), 256, SM_TOTAL, stream>>>(
      inp16, wt[1], bx[1], wpt[1], bp[1], par0, gebuf, out1, par1, 8);
  tree_level_kernel<<<dim3(BROWS / MT, 64), 256, SM_TOTAL, stream>>>(
      inp16, wt[2], bx[2], wpt[2], bp[2], par1, gebuf, out2, nullptr, 64);
}
